// AgentAttention_14173392077278
// MI455X (gfx1250) — matrix operand order repaired and checked by reading, not compiled, not run on hardware
//
#include <hip/hip_runtime.h>
#include <hip/hip_bf16.h>
#include <math.h>
#include <stdint.h>

// ---------------- problem constants ----------------
#define BB   16
#define NN   3136          // 56*56
#define CC   256
#define NHH  8
#define DD   32            // head dim
#define AGG  49            // agents (7x7)
#define HWW  56
#define MM   (BB*NN)       // 50176 rows
#define KT   8             // K tiles of 32 (K = 256)
#define SCALE 0.17677669529663687f   // 32^-0.5

typedef __attribute__((ext_vector_type(16))) _Float16 v16h;
typedef __attribute__((ext_vector_type(8)))  float    v8f;

static __device__ __forceinline__ int a_kmap(int j, int half) {
    // contiguous k order: k = 16*half + j (matches B fills)
    return 16 * half + j;
}

// ======================================================================
// Pack f32 row-major A [M x 256] into WMMA A-fragment order f16:
// Ap[mt][kt][lane][16], one thread per packed element.
// ======================================================================
__global__ void pack_a(const float* __restrict__ A, _Float16* __restrict__ Ap)
{
    int t = blockIdx.x * blockDim.x + threadIdx.x;    // M*256 threads
    int j    = t & 15;
    int lane = (t >> 4) & 31;
    int kt   = (t >> 9) & 7;
    int mt   = t >> 12;
    int m = mt * 16 + (lane & 15);
    int k = kt * 32 + a_kmap(j, lane >> 4);
    Ap[t] = (_Float16)A[(size_t)m * CC + k];
}

// ======================================================================
// Pack f32 weights [256 x Ncols(ldb)] into WMMA B-fragment order f16:
// Bp[tn][kt][lane][16]  (tn = 16-wide column tile, 0..15)
// ======================================================================
__global__ void pack_b(const float* __restrict__ Bw, int ldb, _Float16* __restrict__ Bp)
{
    int t = blockIdx.x * blockDim.x + threadIdx.x;    // 65536 threads
    int j    = t & 15;
    int lane = (t >> 4) & 31;
    int kt   = (t >> 9) & 7;
    int tn   = t >> 12;
    int n = tn * 16 + (lane & 15);
    int k = kt * 32 + 16 * (lane >> 4) + j;
    Bp[t] = (_Float16)Bw[(size_t)k * ldb + n];
}

// ======================================================================
// Packed-f16 GEMM: C[M x 256] = A(pack) * B(pack) (+bias), f32 out.
// Block = 8 waves = 128 rows x 64 cols. The 32KB packed-B panel shared by
// the whole block is staged into LDS with CDNA5 async loads, then each
// wave streams A fragments (b128 pairs) and issues 4 WMMAs per K-step.
// ======================================================================
__global__ void gemm_packed(const _Float16* __restrict__ Ap,
                            const _Float16* __restrict__ Bp,
                            float* __restrict__ Cm,
                            const float* __restrict__ bias)
{
    __shared__ _Float16 bs[4 * KT * 32 * 16];         // 16384 f16 = 32 KB

    int bn = blockIdx.x & 3;                          // 64-col panel
    int bm = blockIdx.x >> 2;                         // 128-row panel
    int tid = threadIdx.x;

    // ---- async copy of the contiguous 32KB B panel into LDS ----
    {
        const _Float16* src = Bp + (size_t)bn * 4 * KT * 512;   // 4 n-tiles
        uint32_t lbase = (uint32_t)(uintptr_t)(&bs[0]);
#pragma unroll
        for (int i = 0; i < 8; ++i) {
            int u = tid + i * 256;                    // 16-byte units, 2048 total
            uint64_t g = (uint64_t)(uintptr_t)src + (uint64_t)u * 16;
            uint32_t l = lbase + (uint32_t)u * 16;
            asm volatile("global_load_async_to_lds_b128 %0, %1, off"
                         :: "v"(l), "v"(g) : "memory");
        }
        asm volatile("s_wait_asynccnt 0" ::: "memory");
    }
    __syncthreads();

    int wave = tid >> 5, lane = tid & 31;
    int mt = bm * 8 + wave;

    const v16h* bsv = reinterpret_cast<const v16h*>(bs);
    v8f acc[4] = {};
#pragma unroll
    for (int kt = 0; kt < KT; ++kt) {
        v16h af = *reinterpret_cast<const v16h*>(
            Ap + ((((size_t)mt * KT + kt) * 32 + lane) << 4));
        v16h bfr[4];
#pragma unroll
        for (int nf = 0; nf < 4; ++nf)
            bfr[nf] = bsv[(nf * KT + kt) * 32 + lane];
#pragma unroll
        for (int nf = 0; nf < 4; ++nf)
            acc[nf] = __builtin_amdgcn_wmma_f32_16x16x32_f16(
                false, af, false, bfr[nf], (short)0, acc[nf], false, false);
    }

    int half = lane >> 4, l15 = lane & 15;
#pragma unroll
    for (int nf = 0; nf < 4; ++nf) {
        int n = bn * 64 + nf * 16 + l15;
        float bv = bias ? bias[n] : 0.f;
#pragma unroll
        for (int r = 0; r < 8; ++r) {
            int m = mt * 16 + r + 8 * half;
            Cm[(size_t)m * CC + n] = acc[nf][r] + bv;
        }
    }
}

// ======================================================================
// agent pooling: agent_h[b][h][64][32], rows 49..63 zero-padded
// ======================================================================
__global__ void pool_agent(const float* __restrict__ q, float* __restrict__ agent_h)
{
    int t = blockIdx.x * blockDim.x + threadIdx.x;       // 262144 threads
    int d  = t & 31;
    int a  = (t >> 5) & 63;
    int h  = (t >> 11) & 7;
    int b  = t >> 14;
    float s = 0.f;
    if (a < AGG) {
        int ay = a / 7, ax = a % 7;
#pragma unroll
        for (int iy = 0; iy < 8; ++iy)
            for (int ix = 0; ix < 8; ++ix) {
                int y = ay * 8 + iy, x = ax * 8 + ix;
                s += q[((size_t)b * NN + y * HWW + x) * CC + h * DD + d];
            }
        s *= (1.f / 64.f);
    }
    agent_h[t] = s;
}

// ======================================================================
// bilinear 7x7 -> 56x56 (half-pixel centers, edge clamp)
// ======================================================================
static __device__ __forceinline__ float bilerp7(const float* __restrict__ g, int y, int x)
{
    float sy = fminf(fmaxf((y + 0.5f) * 0.125f - 0.5f, 0.f), 6.f);
    float sx = fminf(fmaxf((x + 0.5f) * 0.125f - 0.5f, 0.f), 6.f);
    int y0 = (int)sy; if (y0 > 5) y0 = 5;
    int x0 = (int)sx; if (x0 > 5) x0 = 5;
    float fy = sy - y0, fx = sx - x0;
    float v00 = g[y0 * 7 + x0],       v01 = g[y0 * 7 + x0 + 1];
    float v10 = g[(y0 + 1) * 7 + x0], v11 = g[(y0 + 1) * 7 + x0 + 1];
    float t0 = v00 + fx * (v01 - v00);
    float t1 = v10 + fx * (v11 - v10);
    return t0 + fy * (t1 - t0);
}

__global__ void bias_pb1(const float* __restrict__ an_bias, const float* __restrict__ ah_bias,
                         const float* __restrict__ aw_bias, float* __restrict__ pb1)
{
    int t = blockIdx.x * blockDim.x + threadIdx.x;       // 8*49*3136
    int n = t % NN;
    int a = (t / NN) % AGG;
    int h = t / (NN * AGG);
    int y = n / HWW, x = n % HWW;
    float v = bilerp7(an_bias + (size_t)(h * AGG + a) * 49, y, x)
            + ah_bias[(h * AGG + a) * HWW + y]
            + aw_bias[(h * AGG + a) * HWW + x];
    pb1[t] = v;
}

__global__ void bias_ab2(const float* __restrict__ na_bias, const float* __restrict__ ha_bias,
                         const float* __restrict__ wa_bias, float* __restrict__ ab2)
{
    int t = blockIdx.x * blockDim.x + threadIdx.x;       // 8*3136*49
    int a = t % AGG;
    int n = (t / AGG) % NN;
    int h = t / (AGG * NN);
    int y = n / HWW, x = n % HWW;
    float v = bilerp7(na_bias + (size_t)(h * AGG + a) * 49, y, x)
            + ha_bias[(h * HWW + y) * AGG + a]
            + wa_bias[(h * HWW + x) * AGG + a];
    ab2[t] = v;
}

// ======================================================================
// S1[bh][a][n] = scale*(agent_h . k_h^T) + pb1    (only a<49 stored)
// ======================================================================
__global__ void attn_scores1(const float* __restrict__ agent_h, const float* __restrict__ kbuf,
                             const float* __restrict__ pb1, float* __restrict__ S)
{
    int wave = (blockIdx.x * blockDim.x + threadIdx.x) >> 5;
    int lane = threadIdx.x & 31;
    int bh = wave / 784, rem = wave % 784;
    int mt = rem / 196, nt = rem % 196;
    int b = bh >> 3, h = bh & 7;
    int m0 = mt << 4, n0 = nt << 4;
    int half = lane >> 4, l15 = lane & 15;

    v16h af, bf;
    const float* arow = agent_h + (size_t)(bh * 64 + m0 + l15) * DD;
#pragma unroll
    for (int j = 0; j < 16; ++j) af[j] = (_Float16)arow[a_kmap(j, half)];
    const float* bcol = kbuf + ((size_t)b * NN + n0 + l15) * CC + h * DD + 16 * half;
#pragma unroll
    for (int j = 0; j < 16; ++j) bf[j] = (_Float16)bcol[j];

    v8f acc = {};
    acc = __builtin_amdgcn_wmma_f32_16x16x32_f16(false, af, false, bf, (short)0, acc, false, false);

#pragma unroll
    for (int r = 0; r < 8; ++r) {
        int a = m0 + r + 8 * half;
        if (a < AGG) {
            int n = n0 + l15;
            S[((size_t)bh * AGG + a) * NN + n] =
                SCALE * acc[r] + pb1[((size_t)h * AGG + a) * NN + n];
        }
    }
}

// ======================================================================
// softmax over rows of length N=3136 (rows = 6272), in place
// ======================================================================
__global__ void softmax_rows_long(float* __restrict__ S)
{
    float* p = S + (size_t)blockIdx.x * NN;
    __shared__ float red[256];
    int t = threadIdx.x;
    float vals[13];
    float mx = -1e30f;
#pragma unroll
    for (int i = 0; i < 13; ++i) {
        int n = t + i * 256;
        if (n < NN) { vals[i] = p[n]; mx = fmaxf(mx, vals[i]); }
        else vals[i] = -1e30f;
    }
    red[t] = mx; __syncthreads();
    for (int s = 128; s > 0; s >>= 1) { if (t < s) red[t] = fmaxf(red[t], red[t + s]); __syncthreads(); }
    mx = red[0]; __syncthreads();
    float sum = 0.f;
#pragma unroll
    for (int i = 0; i < 13; ++i) {
        int n = t + i * 256;
        if (n < NN) { vals[i] = __expf(vals[i] - mx); sum += vals[i]; }
    }
    red[t] = sum; __syncthreads();
    for (int s = 128; s > 0; s >>= 1) { if (t < s) red[t] += red[t + s]; __syncthreads(); }
    float inv = 1.f / red[0];
#pragma unroll
    for (int i = 0; i < 13; ++i) {
        int n = t + i * 256;
        if (n < NN) p[n] = vals[i] * inv;
    }
}

// ======================================================================
// agv[bh][64][32] = P1(49xN, zero-padded to 64) @ v_h(Nx32)
// ======================================================================
__global__ void attn_agv(const float* __restrict__ S, const float* __restrict__ vbuf,
                         float* __restrict__ agv)
{
    int wave = (blockIdx.x * blockDim.x + threadIdx.x) >> 5;
    int lane = threadIdx.x & 31;
    int bh = wave >> 3, tt = wave & 7;
    int mt = tt >> 1, nt = tt & 1;
    int b = bh >> 3, h = bh & 7;
    int m0 = mt << 4, n0 = nt << 4;
    int half = lane >> 4, l15 = lane & 15;

    int a = m0 + l15;
    float amask = (a < AGG) ? 1.f : 0.f;
    int ac = (a < AGG) ? a : (AGG - 1);
    const float* arowbase = S + ((size_t)bh * AGG + ac) * NN;

    v8f acc = {};
    for (int k0 = 0; k0 < NN; k0 += 32) {
        v16h af, bf;
#pragma unroll
        for (int j = 0; j < 16; ++j)
            af[j] = (_Float16)(amask * arowbase[k0 + a_kmap(j, half)]);
        const float* bbase = vbuf + ((size_t)b * NN + k0 + 16 * half) * CC + h * DD + n0 + l15;
#pragma unroll
        for (int j = 0; j < 16; ++j) bf[j] = (_Float16)bbase[(size_t)j * CC];
        acc = __builtin_amdgcn_wmma_f32_16x16x32_f16(false, af, false, bf, (short)0, acc, false, false);
    }
#pragma unroll
    for (int r = 0; r < 8; ++r)
        agv[(size_t)(bh * 64 + m0 + r + 8 * half) * DD + n0 + l15] = acc[r];
}

// ======================================================================
// S2[bh][n][a] = scale*(q_h . agent_h^T) + ab2  (cols a<49 stored)
// ======================================================================
__global__ void attn_scores2(const float* __restrict__ qbuf, const float* __restrict__ agent_h,
                             const float* __restrict__ ab2, float* __restrict__ S)
{
    int wave = (blockIdx.x * blockDim.x + threadIdx.x) >> 5;
    int lane = threadIdx.x & 31;
    int bh = wave / 784, rem = wave % 784;
    int mt = rem / 4, nt = rem % 4;
    int b = bh >> 3, h = bh & 7;
    int m0 = mt << 4, n0 = nt << 4;
    int half = lane >> 4, l15 = lane & 15;

    v16h af, bf;
    const float* arow = qbuf + ((size_t)b * NN + m0 + l15) * CC + h * DD;
#pragma unroll
    for (int j = 0; j < 16; ++j) af[j] = (_Float16)arow[a_kmap(j, half)];
    const float* bcol = agent_h + (size_t)(bh * 64 + n0 + l15) * DD + 16 * half;
#pragma unroll
    for (int j = 0; j < 16; ++j) bf[j] = (_Float16)bcol[j];

    v8f acc = {};
    acc = __builtin_amdgcn_wmma_f32_16x16x32_f16(false, af, false, bf, (short)0, acc, false, false);

#pragma unroll
    for (int r = 0; r < 8; ++r) {
        int a = n0 + l15;
        if (a < AGG) {
            int n = m0 + r + 8 * half;
            S[((size_t)bh * NN + n) * AGG + a] =
                SCALE * acc[r] + ab2[((size_t)h * NN + n) * AGG + a];
        }
    }
}

// ======================================================================
// softmax over rows of length 49 (rows = 401408), one wave/row
// ======================================================================
__global__ void softmax_rows_49(float* __restrict__ S)
{
    int gid = blockIdx.x * blockDim.x + threadIdx.x;
    int wid = gid >> 5, lane = gid & 31;
    float* p = S + (size_t)wid * AGG;
    float s0 = p[lane];
    bool v1 = (lane + 32) < AGG;
    float s1 = v1 ? p[lane + 32] : -1e30f;
    float mx = fmaxf(s0, s1);
#pragma unroll
    for (int m = 16; m > 0; m >>= 1) mx = fmaxf(mx, __shfl_xor(mx, m));
    s0 = __expf(s0 - mx);
    s1 = v1 ? __expf(s1 - mx) : 0.f;
    float sum = s0 + s1;
#pragma unroll
    for (int m = 16; m > 0; m >>= 1) sum += __shfl_xor(sum, m);
    float inv = 1.f / sum;
    p[lane] = s0 * inv;
    if (v1) p[lane + 32] = s1 * inv;
}

// ======================================================================
// xsum[b][n][c] = P2(Nx49 pad 64) @ agv(64x32)
// ======================================================================
__global__ void attn_xout(const float* __restrict__ S, const float* __restrict__ agv,
                          float* __restrict__ xsum)
{
    int wave = (blockIdx.x * blockDim.x + threadIdx.x) >> 5;
    int lane = threadIdx.x & 31;
    int bh = wave / 392, rem = wave % 392;
    int mt = rem >> 1, nt = rem & 1;
    int b = bh >> 3, h = bh & 7;
    int m0 = mt << 4, n0 = nt << 4;
    int half = lane >> 4, l15 = lane & 15;

    const float* prow = S + ((size_t)bh * NN + m0 + l15) * AGG;
    v8f acc = {};
#pragma unroll
    for (int k0 = 0; k0 < 64; k0 += 32) {
        v16h af, bf;
#pragma unroll
        for (int j = 0; j < 16; ++j) {
            int ai = k0 + a_kmap(j, half);
            float mask = (ai < AGG) ? 1.f : 0.f;
            int aic = (ai < AGG) ? ai : (AGG - 1);
            af[j] = (_Float16)(mask * prow[aic]);
        }
        const float* bbase = agv + (size_t)(bh * 64 + k0 + 16 * half) * DD + n0 + l15;
#pragma unroll
        for (int j = 0; j < 16; ++j) bf[j] = (_Float16)bbase[j * DD];
        acc = __builtin_amdgcn_wmma_f32_16x16x32_f16(false, af, false, bf, (short)0, acc, false, false);
    }
#pragma unroll
    for (int r = 0; r < 8; ++r)
        xsum[((size_t)b * NN + m0 + r + 8 * half) * CC + h * DD + n0 + l15] = acc[r];
}

// ======================================================================
// xsum += depthwise 3x3 conv of v image + dwc_b
// ======================================================================
__global__ void dwc_add(const float* __restrict__ vbuf, const float* __restrict__ w,
                        const float* __restrict__ wb, float* __restrict__ xsum)
{
    int t = blockIdx.x * blockDim.x + threadIdx.x;       // B*N*C
    int c = t % CC;
    int n = (t / CC) % NN;
    int b = t / (CC * NN);
    int y = n / HWW, x = n % HWW;
    float acc = wb[c];
#pragma unroll
    for (int dy = -1; dy <= 1; ++dy) {
#pragma unroll
        for (int dx = -1; dx <= 1; ++dx) {
            int yy = y + dy, xx = x + dx;
            if (yy >= 0 && yy < HWW && xx >= 0 && xx < HWW)
                acc += w[c * 9 + (dy + 1) * 3 + (dx + 1)] *
                       vbuf[((size_t)b * NN + yy * HWW + xx) * CC + c];
        }
    }
    xsum[t] += acc;
}

// ======================================================================
extern "C" void kernel_launch(void* const* d_in, const int* in_sizes, int n_in,
                              void* d_out, int out_size, void* d_ws, size_t ws_size,
                              hipStream_t stream)
{
    const float* x_f    = (const float*)d_in[0];
    const float* x_t    = (const float*)d_in[1];
    const float* Wq     = (const float*)d_in[2];
    const float* Wkv    = (const float*)d_in[3];
    const float* Wproj  = (const float*)d_in[4];
    const float* bproj  = (const float*)d_in[5];
    const float* dwc_w  = (const float*)d_in[6];
    const float* dwc_b  = (const float*)d_in[7];
    const float* an_b   = (const float*)d_in[8];
    const float* na_b   = (const float*)d_in[9];
    const float* ah_b   = (const float*)d_in[10];
    const float* aw_b   = (const float*)d_in[11];
    const float* ha_b   = (const float*)d_in[12];
    const float* wa_b   = (const float*)d_in[13];
    float* out = (float*)d_out;

    // ---- workspace layout ----
    float* ws = (float*)d_ws;
    const size_t SZ_QKV = (size_t)MM * CC;              // 12,845,056 f32
    float* q       = ws;
    float* kbuf    = q + SZ_QKV;                        // k, later reused as xsum
    float* vbuf    = kbuf + SZ_QKV;
    float* agent_h = vbuf + SZ_QKV;                     // B*NH*64*32
    float* agv     = agent_h + (size_t)BB * NHH * 64 * DD;
    float* pb1     = agv + (size_t)BB * NHH * 64 * DD;
    float* ab2     = pb1 + (size_t)NHH * AGG * NN;
    float* S       = ab2 + (size_t)NHH * NN * AGG;      // 19,668,992 f32 (S1 then S2)
    float* xsum    = kbuf;                              // alias: k dead after scores1

    _Float16* f16b    = (_Float16*)(S + (size_t)BB * NHH * AGG * NN);
    _Float16* xf_pack = f16b;                           // also reused for xsum pack
    _Float16* xt_pack = xf_pack + SZ_QKV;
    _Float16* Bq_pack = xt_pack + SZ_QKV;
    _Float16* Bk_pack = Bq_pack + 65536;
    _Float16* Bv_pack = Bk_pack + 65536;
    _Float16* Bp_pack = Bv_pack + 65536;

    dim3 blk(256);

    // weight packs (tiny)
    pack_b<<<256, blk, 0, stream>>>(Wq,        CC,     Bq_pack);
    pack_b<<<256, blk, 0, stream>>>(Wkv,       2 * CC, Bk_pack);
    pack_b<<<256, blk, 0, stream>>>(Wkv + CC,  2 * CC, Bv_pack);
    pack_b<<<256, blk, 0, stream>>>(Wproj,     CC,     Bp_pack);

    // activation packs
    pack_a<<<50176, blk, 0, stream>>>(x_f, xf_pack);
    pack_a<<<50176, blk, 0, stream>>>(x_t, xt_pack);

    // q / k / v GEMMs: 392 m-panels x 4 n-panels = 1568 blocks
    gemm_packed<<<1568, blk, 0, stream>>>(xf_pack, Bq_pack, q,    nullptr);
    gemm_packed<<<1568, blk, 0, stream>>>(xt_pack, Bk_pack, kbuf, nullptr);
    gemm_packed<<<1568, blk, 0, stream>>>(xt_pack, Bv_pack, vbuf, nullptr);

    // agent pooling + biases
    pool_agent<<<1024, blk, 0, stream>>>(q, agent_h);
    bias_pb1<<<4802, blk, 0, stream>>>(an_b, ah_b, aw_b, pb1);
    bias_ab2<<<4802, blk, 0, stream>>>(na_b, ha_b, wa_b, ab2);

    // agent attention
    attn_scores1<<<12544, blk, 0, stream>>>(agent_h, kbuf, pb1, S);
    softmax_rows_long<<<6272, blk, 0, stream>>>(S);
    attn_agv<<<128, blk, 0, stream>>>(S, vbuf, agv);

    // query attention (S buffer reused; stream order guarantees safety)
    attn_scores2<<<12544, blk, 0, stream>>>(q, agent_h, ab2, S);
    softmax_rows_49<<<50176, blk, 0, stream>>>(S);
    attn_xout<<<6272, blk, 0, stream>>>(S, agv, xsum);

    // depthwise conv add
    dwc_add<<<50176, blk, 0, stream>>>(vbuf, dwc_w, dwc_b, xsum);

    // final projection: pack xsum (xf_pack region is dead) then packed GEMM
    pack_a<<<50176, blk, 0, stream>>>(xsum, xf_pack);
    gemm_packed<<<1568, blk, 0, stream>>>(xf_pack, Bp_pack, out, bproj);
}
